// Decoder_14783277432881
// MI455X (gfx1250) — matrix operand order repaired and checked by reading, not compiled, not run on hardware
//
#include <hip/hip_runtime.h>
#include <hip/hip_bf16.h>

// ---------------------------------------------------------------------------
// Problem constants (match reference)
// ---------------------------------------------------------------------------
#define BB   64
#define LL   196
#define DD   512
#define HH   512
#define EE   512
#define VV   32000
#define TT   31          // TCAP-1 decode steps
#define GATE (4*HH)      // 2048
#define KCAT (EE+DD+HH)  // 1536

typedef __attribute__((ext_vector_type(16))) _Float16 v16h;
typedef __attribute__((ext_vector_type(8)))  float    v8f;
typedef __attribute__((ext_vector_type(8)))  _Float16 h8;

// ---------------------------------------------------------------------------
// CDNA5 async copy: global -> LDS, 128-bit per lane, tracked by ASYNCcnt.
// VDST operand = per-lane LDS byte offset (addrspace(3) pointer value),
// VADDR = 64-bit global address.
// ---------------------------------------------------------------------------
__device__ __forceinline__ void async_copy_b128(_Float16* lds_dst,
                                                const _Float16* gsrc)
{
    auto ldsp = (__attribute__((address_space(3))) _Float16*)lds_dst;
    asm volatile("global_load_async_to_lds_b128 %0, %1, off"
                 :: "v"(ldsp), "v"(gsrc) : "memory");
}

__device__ __forceinline__ void wait_async0()
{
    asm volatile("s_wait_asynccnt 0x0" ::: "memory");
}

// ---------------------------------------------------------------------------
// Generic f16 WMMA GEMM:  C[M][N] = A[M][K] * Bt[N][K]^T (+ bias[n])
// A row-major [M][K], Bt row-major [N][K] (i.e. torch weight layout for x@W.T)
// Requires: M % 64 == 0, N % 128 == 0, K % 32 == 0.
// Block: 256 threads = 8 waves. Block tile 64(M) x 128(N).
// Wave w: mt = w&3 -> 16 M-rows; npair = w>>2 -> 64 N-cols (4 x 16 subtiles).
// LDS tiles double-buffered; fills via global_load_async_to_lds_b128.
// ---------------------------------------------------------------------------
__global__ __launch_bounds__(256)
void gemm_wmma_f16(const _Float16* __restrict__ A,
                   const _Float16* __restrict__ Bt,
                   float* __restrict__ C, size_t ldc,
                   int K,
                   const float* __restrict__ bias)
{
    __shared__ _Float16 As[2][64 * 32];
    __shared__ _Float16 Bs[2][128 * 32];

    const int tid   = threadIdx.x;
    const int lane  = tid & 31;
    const int w     = tid >> 5;
    const int mt    = w & 3;
    const int npair = w >> 2;
    const int m0    = blockIdx.y * 64;
    const int n0    = blockIdx.x * 128;

    // tile-load assignments
    const int arow = tid >> 2;            // 0..63
    const int akc  = (tid & 3) * 8;       // 0,8,16,24 halves
    const int brow = tid >> 1;            // 0..127
    const int bkc  = (tid & 1) * 16;      // 0 or 16 halves

    const _Float16* gA = A  + (size_t)(m0 + arow) * K + akc;
    const _Float16* gB = Bt + (size_t)(n0 + brow) * K + bkc;

    v8f acc[4] = {v8f{}, v8f{}, v8f{}, v8f{}};

    const int m = lane & 15;
    const int g = lane >> 4;              // half-wave group

    // prefetch tile 0 into buffer 0
    async_copy_b128(&As[0][arow * 32 + akc],     gA);
    async_copy_b128(&Bs[0][brow * 32 + bkc],     gB);
    async_copy_b128(&Bs[0][brow * 32 + bkc + 8], gB + 8);

    int buf = 0;
    for (int kk = 0; kk < K; kk += 32) {
        wait_async0();
        __syncthreads();

        // prefetch next tile into the other buffer (safe after barrier)
        if (kk + 32 < K) {
            const int nb = buf ^ 1;
            async_copy_b128(&As[nb][arow * 32 + akc],     gA + kk + 32);
            async_copy_b128(&Bs[nb][brow * 32 + bkc],     gB + kk + 32);
            async_copy_b128(&Bs[nb][brow * 32 + bkc + 8], gB + kk + 40);
        }

        // A fragment: 16x32 f16, ISA striping (lanes 0-15: K 0-7/16-23, 16-31: +8)
        v16h afrag;
        const _Float16* ap = &As[buf][(mt * 16 + m) * 32];
        #pragma unroll
        for (int j = 0; j < 8; ++j) {
            int k0 = ((j & 3) << 1) + (g << 3) + ((j >> 2) << 4);
            afrag[2 * j]     = ap[k0];
            afrag[2 * j + 1] = ap[k0 + 1];
        }

        #pragma unroll
        for (int s = 0; s < 4; ++s) {
            // B fragment: 32x16 f16, same striping as A (lanes 0-15: K 0-7/16-23, 16-31: +8)
            const _Float16* bp = &Bs[buf][(npair * 64 + s * 16 + m) * 32];
            v16h bfrag;
            #pragma unroll
            for (int j = 0; j < 8; ++j) {
                int k0 = ((j & 3) << 1) + (g << 3) + ((j >> 2) << 4);
                bfrag[2 * j]     = bp[k0];
                bfrag[2 * j + 1] = bp[k0 + 1];
            }
            acc[s] = __builtin_amdgcn_wmma_f32_16x16x32_f16(
                false, afrag, false, bfrag, (short)0, acc[s], false, false);
        }
        buf ^= 1;
    }

    // D layout: col = lane&15 ; row = vgpr r + 8*(lane>>4)
    const int col_base = n0 + npair * 64 + m;
    const int row_base = m0 + mt * 16 + (g << 3);
    #pragma unroll
    for (int s = 0; s < 4; ++s) {
        const int col = col_base + s * 16;
        const float bv = bias ? bias[col] : 0.0f;
        #pragma unroll
        for (int r = 0; r < 8; ++r) {
            C[(size_t)(row_base + r) * ldc + col] = acc[s][r] + bv;
        }
    }
}

// ---------------------------------------------------------------------------
// h0 = c0 = mean over L of img_features
// ---------------------------------------------------------------------------
__global__ void init_state_kernel(const float* __restrict__ img,
                                  float* __restrict__ h, float* __restrict__ c,
                                  _Float16* __restrict__ h_h)
{
    int idx = blockIdx.x * blockDim.x + threadIdx.x;   // 64*512
    int b = idx >> 9, d = idx & 511;
    const float* p = img + (size_t)b * LL * DD + d;
    float s = 0.0f;
    for (int l = 0; l < LL; ++l) s += p[(size_t)l * DD];
    s *= (1.0f / (float)LL);
    h[idx] = s; c[idx] = s; h_h[idx] = (_Float16)s;
}

// ---------------------------------------------------------------------------
// elementwise helpers
// ---------------------------------------------------------------------------
__global__ void cast_f32_to_f16_kernel(const float* __restrict__ src,
                                       _Float16* __restrict__ dst, long n)
{
    long i = (long)blockIdx.x * blockDim.x + threadIdx.x;
    long stride = (long)gridDim.x * blockDim.x;
    for (; i < n; i += stride) dst[i] = (_Float16)src[i];
}

__global__ void transpose_cast_kernel(const float* __restrict__ src,
                                      _Float16* __restrict__ dst,
                                      int rows, int cols)
{
    long i = (long)blockIdx.x * blockDim.x + threadIdx.x;
    long n = (long)rows * cols;
    long stride = (long)gridDim.x * blockDim.x;
    for (; i < n; i += stride) {
        int r = (int)(i / cols), c = (int)(i % cols);
        dst[(size_t)c * rows + r] = (_Float16)src[i];
    }
}

__global__ void concat_w_kernel(const float* __restrict__ W_ih,
                                const float* __restrict__ W_hh,
                                _Float16* __restrict__ Wcat)
{
    long i = (long)blockIdx.x * blockDim.x + threadIdx.x;   // GATE*KCAT
    int row = (int)(i / KCAT), col = (int)(i % KCAT);
    float v = (col < (EE + DD)) ? W_ih[(size_t)row * (EE + DD) + col]
                                : W_hh[(size_t)row * HH + (col - (EE + DD))];
    Wcat[i] = (_Float16)v;
}

__global__ void bias_cat_kernel(const float* __restrict__ b_ih,
                                const float* __restrict__ b_hh,
                                float* __restrict__ bc)
{
    int i = blockIdx.x * blockDim.x + threadIdx.x;   // 2048
    bc[i] = b_ih[i] + b_hh[i];
}

__global__ void gather_emb_kernel(const int* __restrict__ captions,
                                  const float* __restrict__ table,
                                  _Float16* __restrict__ emb)
{
    long i = (long)blockIdx.x * blockDim.x + threadIdx.x;  // TT*BB*EE
    int e = (int)(i & 511);
    int b = (int)((i >> 9) & 63);
    int t = (int)(i >> 15);
    int tok = captions[b * (TT + 1) + t];
    emb[i] = (_Float16)table[(size_t)tok * EE + e];
}

__global__ void build_x_kernel(const _Float16* __restrict__ emb_t,
                               const float* __restrict__ context,
                               const _Float16* __restrict__ h_h,
                               _Float16* __restrict__ x)
{
    int i = blockIdx.x * blockDim.x + threadIdx.x;   // 64*1536
    int b = i / KCAT, col = i % KCAT;
    _Float16 v;
    if (col < EE)            v = emb_t[b * EE + col];
    else if (col < EE + DD)  v = (_Float16)context[b * DD + (col - EE)];
    else                     v = h_h[b * HH + (col - EE - DD)];
    x[i] = v;
}

// ---------------------------------------------------------------------------
// Bahdanau attention: scores -> softmax -> alpha, context.  Block per b.
// ---------------------------------------------------------------------------
__global__ __launch_bounds__(256)
void attention_kernel(const float* __restrict__ feat_proj,
                      const float* __restrict__ whh,
                      const float* __restrict__ att_bias,
                      const float* __restrict__ v_att,
                      const float* __restrict__ img,
                      float* __restrict__ alpha_out, long alpha_bstride,
                      float* __restrict__ context)
{
    const int b = blockIdx.x;
    const int tid = threadIdx.x;
    const int lane = tid & 31, w = tid >> 5;

    __shared__ float wh[DD];
    __shared__ float va[DD];
    __shared__ float sc[256];
    __shared__ float red[256];

    for (int d = tid; d < DD; d += 256) {
        wh[d] = whh[b * DD + d];
        va[d] = v_att[d];
    }
    __syncthreads();

    // scores: one attention location per wave, strided
    for (int l = w; l < LL; l += 8) {
        const float* fp = feat_proj + ((size_t)b * LL + l) * DD;
        const float ab = att_bias[l];
        float partial = 0.0f;
        for (int d = lane; d < DD; d += 32) {
            float v = fp[d] + wh[d] + ab;
            v = v > 0.0f ? v : 0.0f;
            partial += v * va[d];
        }
        #pragma unroll
        for (int off = 16; off > 0; off >>= 1)
            partial += __shfl_xor(partial, off, 32);
        if (lane == 0) sc[l] = partial;
    }
    __syncthreads();

    // softmax over L (196, padded to 256)
    float myv = (tid < LL) ? sc[tid] : -1e30f;
    red[tid] = myv; __syncthreads();
    for (int s = 128; s > 0; s >>= 1) {
        if (tid < s) red[tid] = fmaxf(red[tid], red[tid + s]);
        __syncthreads();
    }
    float mx = red[0];
    __syncthreads();
    float e = (tid < LL) ? __expf(myv - mx) : 0.0f;
    red[tid] = e; __syncthreads();
    for (int s = 128; s > 0; s >>= 1) {
        if (tid < s) red[tid] += red[tid + s];
        __syncthreads();
    }
    float inv = 1.0f / red[0];
    __syncthreads();
    if (tid < LL) {
        float a = e * inv;
        sc[tid] = a;
        alpha_out[(size_t)b * alpha_bstride + tid] = a;
    }
    __syncthreads();

    // context[b][d] = sum_l alpha[l] * img[b][l][d]
    for (int d = tid; d < DD; d += 256) {
        const float* ib = img + (size_t)b * LL * DD + d;
        float acc = 0.0f;
        for (int l = 0; l < LL; ++l) acc += sc[l] * ib[(size_t)l * DD];
        context[b * DD + d] = acc;
    }
}

// ---------------------------------------------------------------------------
// LSTM cell pointwise (torch gate order i,f,g,o)
// ---------------------------------------------------------------------------
__global__ void lstm_cell_kernel(const float* __restrict__ gates,
                                 float* __restrict__ h, float* __restrict__ c,
                                 _Float16* __restrict__ h_h)
{
    int idx = blockIdx.x * blockDim.x + threadIdx.x;   // 64*512
    int b = idx >> 9, j = idx & 511;
    const float* gb = gates + (size_t)b * GATE;
    float gi = gb[j], gf = gb[HH + j], gg = gb[2 * HH + j], go = gb[3 * HH + j];
    float si = 1.0f / (1.0f + __expf(-gi));
    float sf = 1.0f / (1.0f + __expf(-gf));
    float so = 1.0f / (1.0f + __expf(-go));
    float tg = tanhf(gg);
    float cn = sf * c[idx] + si * tg;
    float hn = so * tanhf(cn);
    c[idx] = cn;
    h[idx] = hn;
    h_h[idx] = (_Float16)hn;
}

// ---------------------------------------------------------------------------
// host orchestration
// ---------------------------------------------------------------------------
extern "C" void kernel_launch(void* const* d_in, const int* in_sizes, int n_in,
                              void* d_out, int out_size, void* d_ws, size_t ws_size,
                              hipStream_t stream)
{
    const float* img      = (const float*)d_in[0];
    const int*   captions = (const int*)  d_in[1];
    const float* embed    = (const float*)d_in[2];
    const float* W_feat   = (const float*)d_in[3];
    const float* W_h      = (const float*)d_in[4];
    const float* att_bias = (const float*)d_in[5];
    const float* v_att    = (const float*)d_in[6];
    const float* W_ih     = (const float*)d_in[7];
    const float* W_hh     = (const float*)d_in[8];
    const float* b_ih     = (const float*)d_in[9];
    const float* b_hh     = (const float*)d_in[10];
    const float* W_out    = (const float*)d_in[11];
    const float* b_out    = (const float*)d_in[12];

    float* out_pred  = (float*)d_out;                                  // [B,T,V]
    float* out_alpha = (float*)d_out + (size_t)BB * TT * VV;           // [B,T,L]

    // workspace carve-out
    char* ws = (char*)d_ws;
    auto alloc = [&](size_t bytes) -> char* {
        char* p = ws;
        ws += (bytes + 255) & ~(size_t)255;
        return p;
    };
    _Float16* img_h     = (_Float16*)alloc((size_t)BB * LL * DD * 2);
    float*    feat_proj = (float*)   alloc((size_t)BB * LL * DD * 4);
    _Float16* Wfeat_t   = (_Float16*)alloc((size_t)DD * DD * 2);
    _Float16* Wh_t      = (_Float16*)alloc((size_t)HH * DD * 2);
    _Float16* Wcat      = (_Float16*)alloc((size_t)GATE * KCAT * 2);
    float*    bias_cat  = (float*)   alloc((size_t)GATE * 4);
    _Float16* Wout_h    = (_Float16*)alloc((size_t)VV * HH * 2);
    _Float16* emb_h     = (_Float16*)alloc((size_t)TT * BB * EE * 2);
    float*    h_f       = (float*)   alloc((size_t)BB * HH * 4);
    float*    c_f       = (float*)   alloc((size_t)BB * HH * 4);
    _Float16* h_h       = (_Float16*)alloc((size_t)BB * HH * 2);
    float*    whh       = (float*)   alloc((size_t)BB * DD * 4);
    float*    context   = (float*)   alloc((size_t)BB * DD * 4);
    _Float16* x_h       = (_Float16*)alloc((size_t)BB * KCAT * 2);
    float*    gates     = (float*)   alloc((size_t)BB * GATE * 4);

    const int TPB = 256;
    auto blks = [](long n, int tpb) { return (unsigned)((n + tpb - 1) / tpb); };

    // ---- one-time setup ----
    {
        long n = (long)BB * LL * DD;
        cast_f32_to_f16_kernel<<<blks(n, TPB), TPB, 0, stream>>>(img, img_h, n);
    }
    transpose_cast_kernel<<<blks((long)DD * DD, TPB), TPB, 0, stream>>>(W_feat, Wfeat_t, DD, DD);
    transpose_cast_kernel<<<blks((long)HH * DD, TPB), TPB, 0, stream>>>(W_h, Wh_t, HH, DD);
    concat_w_kernel<<<blks((long)GATE * KCAT, TPB), TPB, 0, stream>>>(W_ih, W_hh, Wcat);
    bias_cat_kernel<<<blks(GATE, TPB), TPB, 0, stream>>>(b_ih, b_hh, bias_cat);
    {
        long n = (long)VV * HH;
        cast_f32_to_f16_kernel<<<blks(n, TPB), TPB, 0, stream>>>(W_out, Wout_h, n);
    }
    gather_emb_kernel<<<blks((long)TT * BB * EE, TPB), TPB, 0, stream>>>(captions, embed, emb_h);
    init_state_kernel<<<blks((long)BB * HH, TPB), TPB, 0, stream>>>(img, h_f, c_f, h_h);

    // feat_proj = img @ W_feat   : M=12544, N=512, K=512
    gemm_wmma_f16<<<dim3(DD / 128, (BB * LL) / 64), 256, 0, stream>>>(
        img_h, Wfeat_t, feat_proj, (size_t)DD, DD, nullptr);

    // ---- decode steps ----
    for (int t = 0; t < TT; ++t) {
        // whh = h @ W_h : M=64, N=512, K=512
        gemm_wmma_f16<<<dim3(DD / 128, 1), 256, 0, stream>>>(
            h_h, Wh_t, whh, (size_t)DD, HH, nullptr);

        attention_kernel<<<BB, 256, 0, stream>>>(
            feat_proj, whh, att_bias, v_att, img,
            out_alpha + (size_t)t * LL, (long)TT * LL, context);

        build_x_kernel<<<blks((long)BB * KCAT, TPB), TPB, 0, stream>>>(
            emb_h + (size_t)t * BB * EE, context, h_h, x_h);

        // gates = x @ [W_ih|W_hh]^T + (b_ih+b_hh) : M=64, N=2048, K=1536
        gemm_wmma_f16<<<dim3(GATE / 128, 1), 256, 0, stream>>>(
            x_h, Wcat, gates, (size_t)GATE, KCAT, bias_cat);

        lstm_cell_kernel<<<blks((long)BB * HH, TPB), TPB, 0, stream>>>(
            gates, h_f, c_f, h_h);

        // logits = h_new @ W_out^T + b_out : M=64, N=32000, K=512
        gemm_wmma_f16<<<dim3(VV / 128, 1), 256, 0, stream>>>(
            h_h, Wout_h, out_pred + (size_t)t * VV, (size_t)TT * VV, HH, b_out);
    }
}
